// ContinuousCWTLayer_20882130993234
// MI455X (gfx1250) — matrix operand order repaired and checked by reading, not compiled, not run on hardware
//
#include <hip/hip_runtime.h>
#include <math.h>

// ---------------- problem constants ----------------
#define BB      4
#define CC      16
#define MAXT    2048
#define FD      32
#define TOK     256
#define KTAPS   2049          // MAXT is even -> K = MAXT+1
#define NKB     65            // ceil(2049/32)
#define KPAD    (NKB*32)      // 2080
#define PADL    1056          // >= 1024 + margin, keeps all windows in-bounds
#define XLEN    (PADL + MAXT + 1056)   // 4160 halves per (b,c)

typedef __attribute__((ext_vector_type(16))) _Float16 v16h;
typedef __attribute__((ext_vector_type(8)))  float    v8f;

// workspace layout (halves):
//   xh  : BB*CC*XLEN                      (padded f16 copy of x)
//   wv  : BB * 260 frags * 512 halves     (wavelets, pre-swizzled B-fragments)
// frag index within a batch: (comp*2 + ftile)*NKB + kb ; element = lane*16 + i
#define XH_HALVES  (BB*CC*XLEN)
#define WV_PER_B   (2*2*NKB*32*16)        // 133120 halves per batch
#define PI_F 3.14159265358979323846f

// ---------------- kernel 1: pad + f32->f16 convert x ----------------
__global__ void cwt_prep_x(const float* __restrict__ x, _Float16* __restrict__ xh) {
    int idx = blockIdx.x * 256 + threadIdx.x;
    if (idx >= XH_HALVES) return;
    int t  = idx % XLEN;
    int bc = idx / XLEN;
    int tt = t - PADL;
    float v = 0.f;
    if (tt >= 0 && tt < MAXT) v = x[bc * MAXT + tt];
    xh[idx] = (_Float16)v;
}

// ---------------- kernel 2: build normalized wavelets, pre-swizzled ----------------
// one block per (b,f); 256 threads
__global__ void cwt_build_wavelets(const float* __restrict__ fs,
                                   const float* __restrict__ freqs,
                                   const float* __restrict__ ncyc,
                                   _Float16* __restrict__ wv) {
    __shared__ float s_wr[KPAD];
    __shared__ float s_wi[KPAD];
    __shared__ float s_red[256];

    const int b = blockIdx.x >> 5;     // FD == 32
    const int f = blockIdx.x & 31;
    const int tid = threadIdx.x;

    const float fsb   = fs[b];
    const float fclip = fmaxf(freqs[f], 0.1f);
    const float nc    = fmaxf(ncyc[f], 1.0f);
    const float two_pi_f = 2.0f * PI_F * fclip;
    const float sigma = nc / two_pi_f;
    const float inv2s2 = 1.0f / (2.0f * sigma * sigma);

    float partial = 0.f;
    for (int k = tid; k < KPAD; k += 256) {
        float wr = 0.f, wi = 0.f;
        if (k < KTAPS) {
            float ts  = (float)(k - 1024) / fsb;
            float env = expf(-(ts * ts) * inv2s2);
            float arg = two_pi_f * ts;
            wr = cosf(arg) * env;
            wi = sinf(arg) * env;
            partial += sqrtf(wr * wr + wi * wi);
        }
        s_wr[k] = wr;
        s_wi[k] = wi;
    }
    s_red[tid] = partial;
    __syncthreads();
    for (int s = 128; s > 0; s >>= 1) {
        if (tid < s) s_red[tid] += s_red[tid + s];
        __syncthreads();
    }
    const float inv = 1.0f / (s_red[0] + 1e-8f);

    // scatter into WMMA B-fragment order:
    // B(32x16) layout: lane = ((klocal/8)%2)*16 + (f%16); half i = klocal%8 + (klocal/16)*8; col = f%16
    const int ftile = f >> 4;
    const int fmod  = f & 15;
    _Float16* wvb = wv + (size_t)b * WV_PER_B;
    for (int k = tid; k < KPAD; k += 256) {
        int kb   = k >> 5;
        int kl   = k & 31;
        int lane = (((kl >> 3) & 1) << 4) + fmod;
        int i    = (kl & 7) + ((kl >> 4) << 3);
        int ar   = (((0 * 2 + ftile) * NKB + kb) * 32 + lane) * 16 + i;
        int ai   = (((1 * 2 + ftile) * NKB + kb) * 32 + lane) * 16 + i;
        wvb[ar] = (_Float16)(s_wr[k] * inv);
        wvb[ai] = (_Float16)(s_wi[k] * inv);
    }
}

// ---------------- kernel 3: fused WMMA conv @ sampled positions + interp + mag/phase ----
// one wave32 per (b, token j); 8 waves per block
__global__ void __launch_bounds__(256)
cwt_wmma(const _Float16* __restrict__ xh,
         const _Float16* __restrict__ wv,
         const int* __restrict__ seq_lens,
         float* __restrict__ out) {
    const int wid  = blockIdx.x * 8 + (threadIdx.x >> 5);
    const int lane = threadIdx.x & 31;
    const int b = wid >> 8;        // TOK == 256
    const int j = wid & 255;

    // exact sampling-position math from the reference
    const int   L     = seq_lens[b];
    const float step  = (float)j / (float)(TOK - 1);
    const float end_x = -1.0f + 2.0f * ((float)L - 1.0f) / (float)(MAXT - 1);
    const float gx    = -1.0f + step * (end_x + 1.0f);
    const float px    = (gx + 1.0f) * 0.5f * (float)(MAXT - 1);
    const float fx0   = floorf(px);
    int   t0    = (int)fx0;
    float w1    = px - fx0;
    float w0    = 1.0f - w1;
    const int t1raw = t0 + 1;
    const bool valid1 = (t1raw <= MAXT - 1);
    const int t1 = valid1 ? t1raw : (MAXT - 1);
    const float w1v = valid1 ? w1 : 0.0f;

    // A fragment addressing (16-bit A 16x32, ISA 7.12.2):
    // lane = hi*16 + c ; halves 0-7 -> taps (hi*8 .. hi*8+7), halves 8-15 -> +16
    const int c_lo = lane & 15;
    const int hi   = lane >> 4;
    const _Float16* rowA = xh + ((size_t)(b * CC + c_lo)) * XLEN + (PADL - 1024 + hi * 8);
    const _Float16* wvb  = wv + (size_t)b * WV_PER_B + lane * 16;

    v8f aR00 = {}, aR01 = {}, aI00 = {}, aI01 = {};   // position t0
    v8f aR10 = {}, aR11 = {}, aI10 = {}, aI11 = {};   // position t1

    for (int kb = 0; kb < NKB; ++kb) {
        // A fragments: two contiguous 16B chunks per lane (2-byte aligned)
        v16h a0, a1;
        const _Float16* s0 = rowA + t0 + kb * 32;
        const _Float16* s1 = rowA + t1 + kb * 32;
        __builtin_memcpy(&a0, s0, 16);
        __builtin_memcpy(((char*)&a0) + 16, s0 + 16, 16);
        __builtin_memcpy(&a1, s1, 16);
        __builtin_memcpy(((char*)&a1) + 16, s1 + 16, 16);

        // B fragments: fully aligned 32B per lane (pre-swizzled)
        const v16h bR0 = *(const v16h*)(wvb + (size_t)((0 * NKB + kb) * 512));
        const v16h bR1 = *(const v16h*)(wvb + (size_t)((1 * NKB + kb) * 512));
        const v16h bI0 = *(const v16h*)(wvb + (size_t)((2 * NKB + kb) * 512));
        const v16h bI1 = *(const v16h*)(wvb + (size_t)((3 * NKB + kb) * 512));

        aR00 = __builtin_amdgcn_wmma_f32_16x16x32_f16(false, a0, false, bR0, (short)0, aR00, false, false);
        aR01 = __builtin_amdgcn_wmma_f32_16x16x32_f16(false, a0, false, bR1, (short)0, aR01, false, false);
        aI00 = __builtin_amdgcn_wmma_f32_16x16x32_f16(false, a0, false, bI0, (short)0, aI00, false, false);
        aI01 = __builtin_amdgcn_wmma_f32_16x16x32_f16(false, a0, false, bI1, (short)0, aI01, false, false);
        aR10 = __builtin_amdgcn_wmma_f32_16x16x32_f16(false, a1, false, bR0, (short)0, aR10, false, false);
        aR11 = __builtin_amdgcn_wmma_f32_16x16x32_f16(false, a1, false, bR1, (short)0, aR11, false, false);
        aI10 = __builtin_amdgcn_wmma_f32_16x16x32_f16(false, a1, false, bI0, (short)0, aI10, false, false);
        aI11 = __builtin_amdgcn_wmma_f32_16x16x32_f16(false, a1, false, bI1, (short)0, aI11, false, false);
    }

    // D layout (16x16 f32): lane col n = lane%16 ; VGPR v -> row m = v + 8*(lane/16)
    const int nn = lane & 15;
    #pragma unroll
    for (int ft = 0; ft < 2; ++ft) {
        #pragma unroll
        for (int v = 0; v < 8; ++v) {
            const int c = v + 8 * hi;
            const int f = ft * 16 + nn;
            float r0 = (ft == 0) ? aR00[v] : aR01[v];
            float r1 = (ft == 0) ? aR10[v] : aR11[v];
            float i0 = (ft == 0) ? aI00[v] : aI01[v];
            float i1 = (ft == 0) ? aI10[v] : aI11[v];
            float r  = r0 * w0 + r1 * w1v;
            float im = i0 * w0 + i1 * w1v;
            float mag = sqrtf(r * r + im * im + 1e-8f);
            float ph  = atan2f(im, r) * (1.0f / PI_F);
            // out shape (B, C, 2, F, TOK)
            size_t base = ((((size_t)b * CC + c) * 2 + 0) * FD + f) * TOK + j;
            out[base] = mag;
            out[base + (size_t)FD * TOK] = ph;
        }
    }
}

// ---------------- launch ----------------
extern "C" void kernel_launch(void* const* d_in, const int* in_sizes, int n_in,
                              void* d_out, int out_size, void* d_ws, size_t ws_size,
                              hipStream_t stream) {
    const float* x        = (const float*)d_in[0];
    const float* fs       = (const float*)d_in[1];
    const int*   seq_lens = (const int*)  d_in[2];
    const float* freqs    = (const float*)d_in[3];
    const float* ncyc     = (const float*)d_in[4];
    // d_in[5] = target_time_tokens (== 256, compile-time constant here)

    _Float16* xh = (_Float16*)d_ws;
    _Float16* wvl = (_Float16*)((char*)d_ws + (size_t)XH_HALVES * sizeof(_Float16));
    float* out = (float*)d_out;

    cwt_prep_x<<<(XH_HALVES + 255) / 256, 256, 0, stream>>>(x, xh);
    cwt_build_wavelets<<<BB * FD, 256, 0, stream>>>(fs, freqs, ncyc, wvl);
    cwt_wmma<<<(BB * TOK) / 8, 256, 0, stream>>>(xh, wvl, seq_lens, out);
}
